// QuestionEncoder_45294725103655
// MI455X (gfx1250) — compile-verified, matrix operand order checked by reading, not run on hardware
//
#include <hip/hip_runtime.h>
#include <hip/hip_bf16.h>

#define SEQ   8192
#define DDIM  2048
#define HDIM  2048
#define G3H   (3 * HDIM)
#define NBLK_SCAN 64

typedef __attribute__((ext_vector_type(16))) __bf16 v16bf;
typedef __attribute__((ext_vector_type(8)))  __bf16 v8bf;
typedef __attribute__((ext_vector_type(8)))  float  v8f;

__device__ __forceinline__ float bf_lo(unsigned u) { return __uint_as_float(u << 16); }
__device__ __forceinline__ float bf_hi(unsigned u) { return __uint_as_float(u & 0xFFFF0000u); }

// ---------------------------------------------------------------- fp32 -> bf16 (RNE)
__global__ void gru_f32_to_bf16(const float* __restrict__ in,
                                unsigned short* __restrict__ out, int n)
{
    int i = blockIdx.x * blockDim.x + threadIdx.x;
    if (i < n) {
        unsigned u = __float_as_uint(in[i]);
        out[i] = (unsigned short)((u + 0x7FFFu + ((u >> 16) & 1u)) >> 16);
    }
}

// ---------------------------------------------------------------- Phase 1: xi = x @ W_ih^T + b_ih
// Block = 256 threads = 8 waves, arranged 2(M) x 4(N). Wave macro-tile: 64x32.
// Grid: (8192/128) x (6144/128) = 64 x 48.
__global__ __launch_bounds__(256)
void gru_gemm_xi(const __bf16* __restrict__ A,   // x bf16  [SEQ, DDIM] row-major
                 const __bf16* __restrict__ B,   // W_ih bf16 [3H, DDIM] row-major (rows are output cols)
                 const float*  __restrict__ bias,// b_ih [3H]
                 float* __restrict__ C)          // xi [SEQ, 3H]
{
    const int lane  = threadIdx.x & 31;
    const int wave  = threadIdx.x >> 5;
    const int waveM = wave & 1;
    const int waveN = wave >> 1;

    const int m0 = blockIdx.x * 128 + waveM * 64;   // 4 M-tiles of 16
    const int n0 = blockIdx.y * 128 + waveN * 32;   // 2 N-tiles of 16

    const int half = lane >> 4;   // K-half select per ISA 16-bit A/B layout
    const int l16  = lane & 15;

    v8f acc[4][2];
#pragma unroll
    for (int mi = 0; mi < 4; ++mi)
#pragma unroll
        for (int ni = 0; ni < 2; ++ni)
#pragma unroll
            for (int r = 0; r < 8; ++r) acc[mi][ni][r] = 0.0f;

    for (int k0 = 0; k0 < DDIM; k0 += 32) {
        const int ka = k0 + half * 8;          // lanes 0-15: K=k0..k0+7 ; lanes 16-31: K=k0+8..k0+15
        v16bf a[4], b[2];
#pragma unroll
        for (int mi = 0; mi < 4; ++mi) {
            const __bf16* p = A + (size_t)(m0 + mi * 16 + l16) * DDIM + ka;
            v8bf lo = *(const v8bf*)p;         // K block [ka, ka+8)
            v8bf hi = *(const v8bf*)(p + 16);  // K block [ka+16, ka+24)
            a[mi] = __builtin_shufflevector(lo, hi, 0,1,2,3,4,5,6,7,8,9,10,11,12,13,14,15);
            __builtin_prefetch(p + 32, 0, 1);  // next K slab -> global_prefetch_b8
        }
#pragma unroll
        for (int ni = 0; ni < 2; ++ni) {
            const __bf16* p = B + (size_t)(n0 + ni * 16 + l16) * DDIM + ka;
            v8bf lo = *(const v8bf*)p;
            v8bf hi = *(const v8bf*)(p + 16);
            b[ni] = __builtin_shufflevector(lo, hi, 0,1,2,3,4,5,6,7,8,9,10,11,12,13,14,15);
        }
#pragma unroll
        for (int mi = 0; mi < 4; ++mi)
#pragma unroll
            for (int ni = 0; ni < 2; ++ni)
                acc[mi][ni] = __builtin_amdgcn_wmma_f32_16x16x32_bf16(
                    false, a[mi], false, b[ni], (short)0, acc[mi][ni], false, false);
    }

    // C/D layout: VGPR r -> row (r + 8*half), col l16
#pragma unroll
    for (int ni = 0; ni < 2; ++ni) {
        const int col = n0 + ni * 16 + l16;
        const float bv = bias[col];
#pragma unroll
        for (int mi = 0; mi < 4; ++mi) {
            const int rowbase = m0 + mi * 16 + half * 8;
#pragma unroll
            for (int r = 0; r < 8; ++r)
                C[(size_t)(rowbase + r) * G3H + col] = acc[mi][ni][r] + bv;
        }
    }
}

// ---------------------------------------------------------------- Phase 2: persistent sequential scan
// 64 blocks x 256 threads = 512 waves; each wave owns 4 hidden units; W_hh (bf16, 24MB) lives in L2.
__global__ __launch_bounds__(256)
void gru_scan(const float* __restrict__ xi,      // [SEQ, 3H]
              const __bf16* __restrict__ Whh,    // [3H, HDIM] bf16
              const float* __restrict__ bhh,     // [3H]
              float* __restrict__ hbuf,          // 2*HDIM (zeroed)
              unsigned* __restrict__ syncv,      // [0]=count, [64]=generation (zeroed)
              float* __restrict__ out)           // [HDIM]
{
    const int lane   = threadIdx.x & 31;
    const int waveId = blockIdx.x * (blockDim.x >> 5) + (threadIdx.x >> 5); // 0..511

    volatile unsigned* cnt = syncv;
    volatile unsigned* gen = syncv + 64;

    float* h0 = hbuf;
    float* h1 = hbuf + HDIM;

    for (int t = 0; t < SEQ; ++t) {
        const float* hcur  = (t & 1) ? h1 : h0;
        float*       hnext = (t & 1) ? h0 : h1;
        const float* xt    = xi + (size_t)t * G3H;
        const float2* hp   = (const float2*)hcur;

#pragma unroll
        for (int jj = 0; jj < 4; ++jj) {
            const int j = waveId * 4 + jj;
            const __bf16* wr = Whh + (size_t)j * HDIM;
            const __bf16* wz = Whh + (size_t)(j + HDIM) * HDIM;
            const __bf16* wn = Whh + (size_t)(j + 2 * HDIM) * HDIM;

            float sr = 0.0f, sz = 0.0f, sn = 0.0f;
#pragma unroll 4
            for (int it = 0; it < HDIM / 64; ++it) {     // 32 iterations, 2 elements/lane/iter
                const int k  = it * 64 + lane * 2;
                const float2 hv = hp[k >> 1];
                const unsigned ur = *(const unsigned*)(wr + k);
                const unsigned uz = *(const unsigned*)(wz + k);
                const unsigned un = *(const unsigned*)(wn + k);
                sr = fmaf(bf_lo(ur), hv.x, fmaf(bf_hi(ur), hv.y, sr));
                sz = fmaf(bf_lo(uz), hv.x, fmaf(bf_hi(uz), hv.y, sz));
                sn = fmaf(bf_lo(un), hv.x, fmaf(bf_hi(un), hv.y, sn));
            }
#pragma unroll
            for (int off = 16; off; off >>= 1) {
                sr += __shfl_xor(sr, off, 32);
                sz += __shfl_xor(sz, off, 32);
                sn += __shfl_xor(sn, off, 32);
            }
            if (lane == 0) {
                const float hr = sr + bhh[j];
                const float hz = sz + bhh[j + HDIM];
                const float hn = sn + bhh[j + 2 * HDIM];
                const float r = 1.0f / (1.0f + __expf(-(xt[j] + hr)));
                const float z = 1.0f / (1.0f + __expf(-(xt[j + HDIM] + hz)));
                const float n = tanhf(xt[j + 2 * HDIM] + r * hn);
                hnext[j] = (1.0f - z) * n + z * hcur[j];
            }
        }

        // -------- grid-wide barrier (one per timestep) --------
        __syncthreads();
        if (threadIdx.x == 0) {
            __threadfence();
            const unsigned prev = atomicAdd((unsigned*)cnt, 1u);
            if (prev == NBLK_SCAN - 1) {
                *cnt = 0;                 // reset before releasing waiters
                __threadfence();
                atomicAdd((unsigned*)gen, 1u);
            } else {
                while (*gen < (unsigned)(t + 1)) { __builtin_amdgcn_s_sleep(1); }
            }
        }
        __syncthreads();
    }

    // final h lives in h0 (SEQ even: last step wrote h0)
    const float* hf = (SEQ & 1) ? h1 : h0;
    const int gid = blockIdx.x * blockDim.x + threadIdx.x;
    if (gid < HDIM) out[gid] = hf[gid];
}

// ---------------------------------------------------------------- launch
extern "C" void kernel_launch(void* const* d_in, const int* in_sizes, int n_in,
                              void* d_out, int out_size, void* d_ws, size_t ws_size,
                              hipStream_t stream)
{
    const float* x    = (const float*)d_in[0];   // [SEQ, DDIM]
    const float* W_ih = (const float*)d_in[1];   // [3H, DDIM]
    const float* W_hh = (const float*)d_in[2];   // [3H, HDIM]
    const float* b_ih = (const float*)d_in[3];   // [3H]
    const float* b_hh = (const float*)d_in[4];   // [3H]
    float* out = (float*)d_out;

    char* ws = (char*)d_ws;
    size_t off = 0;
    auto take = [&](size_t bytes) { char* p = ws + off; off += (bytes + 255) & ~(size_t)255; return p; };

    float*          xi      = (float*)         take((size_t)SEQ * G3H * sizeof(float));   // 192 MB
    unsigned short* x_bf    = (unsigned short*)take((size_t)SEQ * DDIM * 2);              //  32 MB
    unsigned short* Wih_bf  = (unsigned short*)take((size_t)G3H * DDIM * 2);              //  24 MB
    unsigned short* Whh_bf  = (unsigned short*)take((size_t)G3H * HDIM * 2);              //  24 MB
    char*           hsync   = take(2 * HDIM * sizeof(float) + 512);                       //  h double-buffer + barrier
    float*    hbuf  = (float*)hsync;
    unsigned* syncv = (unsigned*)(hsync + 2 * HDIM * sizeof(float));

    // fp32 -> bf16 conversions
    {
        int n = SEQ * DDIM;
        gru_f32_to_bf16<<<(n + 255) / 256, 256, 0, stream>>>(x, x_bf, n);
        n = G3H * DDIM;
        gru_f32_to_bf16<<<(n + 255) / 256, 256, 0, stream>>>(W_ih, Wih_bf, n);
        n = G3H * HDIM;
        gru_f32_to_bf16<<<(n + 255) / 256, 256, 0, stream>>>(W_hh, Whh_bf, n);
    }

    // Phase 1: WMMA GEMM for all input projections
    dim3 grid(SEQ / 128, G3H / 128);  // 64 x 48
    gru_gemm_xi<<<grid, 256, 0, stream>>>((const __bf16*)x_bf, (const __bf16*)Wih_bf, b_ih, xi);

    // zero h and barrier state (graph-capture legal)
    hipMemsetAsync(hsync, 0, 2 * HDIM * sizeof(float) + 512, stream);

    // Phase 2: persistent recurrent scan
    gru_scan<<<NBLK_SCAN, 256, 0, stream>>>(xi, (const __bf16*)Whh_bf, b_hh, hbuf, syncv, out);
}
